// Net_62225486184967
// MI455X (gfx1250) — matrix operand order repaired and checked by reading, not compiled, not run on hardware
//
#include <hip/hip_runtime.h>
#include <math.h>

// ---------------- problem constants (from reference) ----------------
#define KCONST 5
#define K3     125
#define KB     5          // kernel-index block for bucket workspace

static const int N1 = 12000, N2 = 4000, N3 = 1500, N4 = 600;
static const int E1 = 144000, E2 = 56000, E3 = 24000, E4 = 9600;

typedef _Float16 half_t;
typedef __attribute__((ext_vector_type(16))) _Float16 v16h;
typedef __attribute__((ext_vector_type(8)))  float    v8f;

static inline int ceil_div(int a, int b) { return (a + b - 1) / b; }
static inline int pad16(int n) { return (n + 15) & ~15; }
static inline int pad32(int n) { return (n + 31) & ~31; }

__device__ __forceinline__ float elu_f(float v) { return v > 0.f ? v : (expf(v) - 1.f); }

// monotone order-preserving float<->uint encoding (for atomicMax segment-max)
__device__ __forceinline__ unsigned enc_f32(float f) {
    unsigned u = __float_as_uint(f);
    return (u & 0x80000000u) ? ~u : (u | 0x80000000u);
}
__device__ __forceinline__ float dec_f32(unsigned u) {
    return (u & 0x80000000u) ? __uint_as_float(u & 0x7FFFFFFFu) : __uint_as_float(~u);
}

// ---------------- small utility kernels ----------------
__global__ void zero_f32(float* __restrict__ p, long long n) {
    long long i = (long long)blockIdx.x * blockDim.x + threadIdx.x;
    if (i < n) p[i] = 0.f;
}

__global__ void xpad_kernel(const float* __restrict__ x, float* __restrict__ xp, int Nn) {
    int i = blockIdx.x * blockDim.x + threadIdx.x;
    if (i >= Nn * 32) return;
    xp[i] = ((i & 31) == 0) ? x[i >> 5] : 0.f;
}

__global__ void deg_kernel(const int* __restrict__ dst, float* __restrict__ deg, int E) {
    int e = blockIdx.x * blockDim.x + threadIdx.x;
    if (e < E) atomicAdd(&deg[dst[e]], 1.0f);
}

// degree-1 open B-spline basis: pseudo (E,3) -> basis (E,8), widx (E,8)
__global__ void basis_kernel(const float* __restrict__ pseudo,
                             float* __restrict__ basis, int* __restrict__ widx, int E) {
    int e = blockIdx.x * blockDim.x + threadIdx.x;
    if (e >= E) return;
    float f[3]; int lo[3];
#pragma unroll
    for (int d = 0; d < 3; ++d) {
        float v = pseudo[e * 3 + d] * (float)(KCONST - 1);
        float l = floorf(v);
        l = fminf(fmaxf(l, 0.f), (float)(KCONST - 2));
        f[d]  = v - l;
        lo[d] = (int)l;
    }
#pragma unroll
    for (int s = 0; s < 8; ++s) {
        float b = 1.f; int w = 0; int stride = 1;
#pragma unroll
        for (int d = 0; d < 3; ++d) {
            int bit = (s >> d) & 1;
            b *= bit ? f[d] : (1.f - f[d]);
            w += (lo[d] + bit) * stride;
            stride *= KCONST;
        }
        basis[e * 8 + s] = b;
        widx[e * 8 + s]  = w;
    }
}

// scatter edge messages into per-(node, kernel-in-block) buckets (Cpad-strided, pre-zeroed)
__global__ void scatter_kernel(const float* __restrict__ X, int ldx,
                               const int* __restrict__ src, const int* __restrict__ dst,
                               const float* __restrict__ basis, const int* __restrict__ widx,
                               float* __restrict__ S, int E, int Cin, int Cpad, int k0) {
    long long tid = (long long)blockIdx.x * blockDim.x + threadIdx.x;
    long long total = (long long)E * 8 * Cin;
    if (tid >= total) return;
    int c = (int)(tid % Cin);
    long long r = tid / Cin;
    int j = (int)(r & 7);
    int e = (int)(r >> 3);
    int wi = widx[e * 8 + j];
    if (wi < k0 || wi >= k0 + KB) return;
    float bv = basis[e * 8 + j];
    float xv = X[(size_t)src[e] * ldx + c];
    atomicAdd(&S[((size_t)dst[e] * KB + (wi - k0)) * Cpad + c], bv * xv);
}

// ---------------- weight pre-pack: f32 row-major -> WMMA-swizzled f16 ----------------
// nmat matrices of (Kdim x N). Per matrix, layout (kb, lane, nb, slot):
//   offset = ((kb*32 + lane)*NB16 + nb)*16 + slot
//   k = kb*32 + (slot>>3)*16 + (lane>>4)*8 + (slot&7) ; n = nb*16 + (lane&15)
// so each lane's NB16 fragments for one k-block are contiguous (NB16*32 bytes).
__global__ void pack_b_kernel(const float* __restrict__ B, half_t* __restrict__ Bsw,
                              int Kdim, int N, int KB32, int NB16, int nmat) {
    long long idx = (long long)blockIdx.x * blockDim.x + threadIdx.x;
    long long per = (long long)KB32 * NB16 * 512;
    if (idx >= per * nmat) return;
    int mat = (int)(idx / per);
    long long r = idx % per;
    int slot = (int)(r & 15);
    long long t = r >> 4;
    int nb = (int)(t % NB16); t /= NB16;
    int lane = (int)(t & 31);
    int kb = (int)(t >> 5);
    int k = kb * 32 + (slot >> 3) * 16 + (lane >> 4) * 8 + (slot & 7);
    int n = nb * 16 + (lane & 15);
    float v = (k < Kdim && n < N) ? B[(size_t)mat * Kdim * N + (size_t)k * N + n] : 0.f;
    Bsw[idx] = (half_t)v;
}

// ---------------- WMMA GEMM (padded, unguarded inner loop) ----------------
__device__ __forceinline__ v16h load_a_frag(const float* __restrict__ arow, int k0, int hs) {
    const float4* a4 = (const float4*)(arow + k0 + hs * 8);
    float4 q0 = a4[0], q1 = a4[1], q2 = a4[4], q3 = a4[5];
    v16h a;
    a[0]  = (half_t)q0.x; a[1]  = (half_t)q0.y; a[2]  = (half_t)q0.z; a[3]  = (half_t)q0.w;
    a[4]  = (half_t)q1.x; a[5]  = (half_t)q1.y; a[6]  = (half_t)q1.z; a[7]  = (half_t)q1.w;
    a[8]  = (half_t)q2.x; a[9]  = (half_t)q2.y; a[10] = (half_t)q2.z; a[11] = (half_t)q2.w;
    a[12] = (half_t)q3.x; a[13] = (half_t)q3.y; a[14] = (half_t)q3.z; a[15] = (half_t)q3.w;
    return a;
}

// One wave computes a 16 x (NBW*16) tile: A fragment reused across NBW WMMAs.
// C[m, coloff+n] = opt_elu( A[Mpad x Kpad](lda) @ Bsw + bias ), flags: bit0=bias, bit1=elu
template <int NBW>
__global__ __launch_bounds__(32)
void gemm_kernel(const float* __restrict__ A, int lda, const half_t* __restrict__ Bsw,
                 const float* __restrict__ bias, float* __restrict__ C,
                 int Kpad, int N, int ldc, int coloff, int flags) {
    const int m0 = blockIdx.x * 16;
    const int lane = threadIdx.x;
    const int hs = lane >> 4, lm = lane & 15;
    v8f acc[NBW];
#pragma unroll
    for (int nb = 0; nb < NBW; ++nb) acc[nb] = (v8f){};
    const float* arow = A + (size_t)(m0 + lm) * lda;
    const half_t* bp = Bsw + (size_t)lane * NBW * 16;
    for (int k0 = 0; k0 < Kpad; k0 += 32) {
        __builtin_prefetch(arow + k0 + 32, 0, 1);
        v16h a = load_a_frag(arow, k0, hs);
#pragma unroll
        for (int nb = 0; nb < NBW; ++nb) {
            v16h b = *(const v16h*)(bp + nb * 16);
            acc[nb] = __builtin_amdgcn_wmma_f32_16x16x32_f16(false, a, false, b, (short)0,
                                                             acc[nb], false, false);
        }
        bp += (size_t)NBW * 512;
    }
#pragma unroll
    for (int nb = 0; nb < NBW; ++nb) {
        int n = nb * 16 + lm;
        if (n < N) {
            float bval = (flags & 1) ? bias[n] : 0.f;
#pragma unroll
            for (int r = 0; r < 8; ++r) {
                float v = acc[nb][r] + bval;
                if (flags & 2) v = elu_f(v);
                C[(size_t)(m0 + r + hs * 8) * ldc + coloff + n] = v;   // rows padded in ws
            }
        }
    }
}

// bucket einsum: OUT[n,d] += sum_{kk<KB} S[n,kk,:] @ W[k0+kk]  (Wsw pre-swizzled f16)
template <int NBW>
__global__ __launch_bounds__(32)
void bucket_gemm_kernel(const float* __restrict__ S, const half_t* __restrict__ Wsw,
                        float* __restrict__ OUT, int Cpad, int Cout, int k0) {
    const int m0 = blockIdx.x * 16;
    const int lane = threadIdx.x;
    const int hs = lane >> 4, lm = lane & 15;
    const size_t matstride = (size_t)(Cpad >> 5) * NBW * 512;
    v8f acc[NBW];
#pragma unroll
    for (int nb = 0; nb < NBW; ++nb) acc[nb] = (v8f){};
    for (int kk = 0; kk < KB; ++kk) {
        const float* arow = S + ((size_t)(m0 + lm) * KB + kk) * Cpad;
        const half_t* bp = Wsw + (size_t)(k0 + kk) * matstride + (size_t)lane * NBW * 16;
        for (int c0 = 0; c0 < Cpad; c0 += 32) {
            __builtin_prefetch(arow + c0 + 32, 0, 1);
            v16h a = load_a_frag(arow, c0, hs);
#pragma unroll
            for (int nb = 0; nb < NBW; ++nb) {
                v16h b = *(const v16h*)(bp + nb * 16);
                acc[nb] = __builtin_amdgcn_wmma_f32_16x16x32_f16(false, a, false, b, (short)0,
                                                                 acc[nb], false, false);
            }
            bp += (size_t)NBW * 512;
        }
    }
#pragma unroll
    for (int nb = 0; nb < NBW; ++nb) {
        int n = nb * 16 + lm;
        if (n < Cout) {
#pragma unroll
            for (int r = 0; r < 8; ++r)
                OUT[(size_t)(m0 + r + hs * 8) * Cout + n] += acc[nb][r];  // tile-exclusive
        }
    }
}

// out = elu( acc/max(deg,1) + root_term + bias )
__global__ void finalize_kernel(const float* __restrict__ acc, const float* __restrict__ rootb,
                                const float* __restrict__ bias, const float* __restrict__ deg,
                                float* __restrict__ out, int Nn, int Cout, int ldo, int coloff) {
    int i = blockIdx.x * blockDim.x + threadIdx.x;
    if (i >= Nn * Cout) return;
    int n = i / Cout, d = i % Cout;
    float dg = fmaxf(deg[n], 1.f);
    float v = acc[i] / dg + rootb[i] + bias[d];
    out[(size_t)n * ldo + coloff + d] = elu_f(v);
}

// ---------------- pooling / gather / softmax ----------------
__global__ void pool_init(unsigned* __restrict__ enc, long long n) {
    long long i = (long long)blockIdx.x * blockDim.x + threadIdx.x;
    if (i < n) enc[i] = enc_f32(-3.4e38f);
}
__global__ void pool_scatter(const float* __restrict__ h, const int* __restrict__ cluster,
                             unsigned* __restrict__ enc, int Nhi, int C) {
    int i = blockIdx.x * blockDim.x + threadIdx.x;
    if (i >= Nhi * C) return;
    int n = i / C, c = i % C;
    atomicMax(&enc[(size_t)cluster[n] * C + c], enc_f32(h[i]));
}
__global__ void pool_decode(const unsigned* __restrict__ enc, float* __restrict__ out,
                            long long n) {
    long long i = (long long)blockIdx.x * blockDim.x + threadIdx.x;
    if (i < n) out[i] = dec_f32(enc[i]);
}
__global__ void gather_rows(const float* __restrict__ in, int ldi, const int* __restrict__ idx,
                            float* __restrict__ out, int Nn, int C, int ldo) {
    int i = blockIdx.x * blockDim.x + threadIdx.x;
    if (i >= Nn * C) return;
    int n = i / C, c = i % C;
    out[(size_t)n * ldo + c] = in[(size_t)idx[n] * ldi + c];
}
__global__ void log_softmax4(const float* __restrict__ in, float* __restrict__ out, int Nn) {
    int n = blockIdx.x * blockDim.x + threadIdx.x;
    if (n >= Nn) return;
    float v[4];
#pragma unroll
    for (int c = 0; c < 4; ++c) v[c] = in[n * 4 + c];
    float m = fmaxf(fmaxf(v[0], v[1]), fmaxf(v[2], v[3]));
    float s = 0.f;
#pragma unroll
    for (int c = 0; c < 4; ++c) s += expf(v[c] - m);
    float l = logf(s);
#pragma unroll
    for (int c = 0; c < 4; ++c) out[n * 4 + c] = v[c] - m - l;
}

// ---------------- host-side dispatch helpers ----------------
static void launch_gemm(hipStream_t stream, int gridm, const float* A, int lda,
                        const half_t* Bsw, const float* bias, float* C,
                        int Kpad, int N, int NB16, int ldc, int coloff, int flags) {
    switch (NB16) {
    case 1: gemm_kernel<1><<<gridm, 32, 0, stream>>>(A, lda, Bsw, bias, C, Kpad, N, ldc, coloff, flags); break;
    case 2: gemm_kernel<2><<<gridm, 32, 0, stream>>>(A, lda, Bsw, bias, C, Kpad, N, ldc, coloff, flags); break;
    default: gemm_kernel<4><<<gridm, 32, 0, stream>>>(A, lda, Bsw, bias, C, Kpad, N, ldc, coloff, flags); break;
    }
}
static void launch_bucket(hipStream_t stream, int gridm, const float* S, const half_t* Wsw,
                          float* OUT, int Cpad, int Cout, int NB16, int k0) {
    switch (NB16) {
    case 1: bucket_gemm_kernel<1><<<gridm, 32, 0, stream>>>(S, Wsw, OUT, Cpad, Cout, k0); break;
    case 2: bucket_gemm_kernel<2><<<gridm, 32, 0, stream>>>(S, Wsw, OUT, Cpad, Cout, k0); break;
    default: bucket_gemm_kernel<4><<<gridm, 32, 0, stream>>>(S, Wsw, OUT, Cpad, Cout, k0); break;
    }
}

static void run_conv(hipStream_t stream, const float* xin, int ldx, int Nn, int Nnp,
                     int Cin, int Cpad, int Cout,
                     const int* src, const int* dst, const float* basis, const int* widx,
                     int E, const float* deg, const half_t* wsw, const half_t* rootsw,
                     const float* bias, float* out, int ldo, int coloff,
                     float* s_ws, float* acc_ws, float* root_ws) {
    int NB = (Cout + 15) / 16;
    zero_f32<<<ceil_div(Nnp * Cout, 256), 256, 0, stream>>>(acc_ws, (long long)Nnp * Cout);
    launch_gemm(stream, Nnp / 16, xin, ldx, rootsw, nullptr, root_ws, Cpad, Cout, NB, Cout, 0, 0);
    for (int k0 = 0; k0 < K3; k0 += KB) {
        long long sn = (long long)Nnp * KB * Cpad;
        zero_f32<<<(int)((sn + 255) / 256), 256, 0, stream>>>(s_ws, sn);
        long long tot = (long long)E * 8 * Cin;
        scatter_kernel<<<(int)((tot + 255) / 256), 256, 0, stream>>>(
            xin, ldx, src, dst, basis, widx, s_ws, E, Cin, Cpad, k0);
        launch_bucket(stream, Nnp / 16, s_ws, wsw, acc_ws, Cpad, Cout, NB, k0);
    }
    finalize_kernel<<<ceil_div(Nn * Cout, 256), 256, 0, stream>>>(
        acc_ws, root_ws, bias, deg, out, Nn, Cout, ldo, coloff);
}

static void run_pool(hipStream_t stream, const float* h, const int* cluster, int Nhi, int Nlo,
                     int C, float* out, unsigned* enc) {
    pool_init<<<ceil_div(Nlo * C, 256), 256, 0, stream>>>(enc, (long long)Nlo * C);
    pool_scatter<<<ceil_div(Nhi * C, 256), 256, 0, stream>>>(h, cluster, enc, Nhi, C);
    pool_decode<<<ceil_div(Nlo * C, 256), 256, 0, stream>>>(enc, out, (long long)Nlo * C);
}

static void pack_b(hipStream_t stream, const float* B, half_t* Bsw, int Kdim, int N, int nmat) {
    int KB32 = pad32(Kdim) / 32, NB16 = (N + 15) / 16;
    long long tot = (long long)KB32 * NB16 * 512 * nmat;
    pack_b_kernel<<<(int)((tot + 255) / 256), 256, 0, stream>>>(B, Bsw, Kdim, N, KB32, NB16, nmat);
}

extern "C" void kernel_launch(void* const* d_in, const int* in_sizes, int n_in,
                              void* d_out, int out_size, void* d_ws, size_t ws_size,
                              hipStream_t stream) {
    const float* x = (const float*)d_in[0];
    const float* pseudo[4] = {(const float*)d_in[1], (const float*)d_in[2],
                              (const float*)d_in[3], (const float*)d_in[4]};
    const float *cw[10], *cr[10], *cb[10];
    for (int i = 0; i < 10; ++i) {
        cw[i] = (const float*)d_in[5 + 3 * i];
        cr[i] = (const float*)d_in[6 + 3 * i];
        cb[i] = (const float*)d_in[7 + 3 * i];
    }
    const float* fc1_w = (const float*)d_in[35]; const float* fc1_b = (const float*)d_in[36];
    const float* fc2_w = (const float*)d_in[37]; const float* fc2_b = (const float*)d_in[38];
    const float* sk_w[3] = {(const float*)d_in[39], (const float*)d_in[41], (const float*)d_in[43]};
    const float* sk_b[3] = {(const float*)d_in[40], (const float*)d_in[42], (const float*)d_in[44]};
    const int* ei[4] = {(const int*)d_in[45], (const int*)d_in[46],
                        (const int*)d_in[47], (const int*)d_in[48]};
    const int* cl[3] = {(const int*)d_in[49], (const int*)d_in[50], (const int*)d_in[51]};
    const int En[4] = {E1, E2, E3, E4};
    const int Nn[4] = {N1, N2, N3, N4};
    const int Nnp[4] = {pad16(N1), pad16(N2), pad16(N3), pad16(N4)};
    // conv geometry: {Cin, Cout} per conv (order c1,c12,c2,c22,c3,c32,c4,c5,c6,c7)
    const int cinA[10]  = {1, 32, 64, 64, 64, 64, 64, 80, 80, 80};
    const int coutA[10] = {32, 64, 64, 64, 64, 64, 64, 64, 64, 64};

    // deterministic bump allocator over d_ws
    size_t off = 0;
    auto alloc = [&](size_t bytes) -> void* {
        off = (off + 255) & ~(size_t)255;
        void* p = (char*)d_ws + off;
        off += bytes;
        return p;
    };
    float* basis[4]; int* widx[4]; float* deg[4];
    for (int l = 0; l < 4; ++l) {
        basis[l] = (float*)alloc((size_t)En[l] * 8 * 4);
        widx[l]  = (int*)alloc((size_t)En[l] * 8 * 4);
        deg[l]   = (float*)alloc((size_t)Nn[l] * 4);
    }
    float* xpad = (float*)alloc((size_t)Nnp[0] * 32 * 4);
    float* h1 = (float*)alloc((size_t)Nnp[0] * 64 * 4);
    float* h2 = (float*)alloc((size_t)Nnp[1] * 64 * 4);
    float* h3 = (float*)alloc((size_t)Nnp[2] * 64 * 4);
    float* h4 = (float*)alloc((size_t)Nnp[3] * 64 * 4);
    float* tmpA = (float*)alloc((size_t)Nnp[0] * 64 * 4);
    float* tmpB = (float*)alloc((size_t)Nnp[0] * 64 * 4);
    float* g3 = (float*)alloc((size_t)Nnp[2] * 96 * 4);
    float* g2 = (float*)alloc((size_t)Nnp[1] * 96 * 4);
    float* g1 = (float*)alloc((size_t)Nnp[0] * 96 * 4);
    float* g3c = (float*)alloc((size_t)Nnp[2] * 64 * 4);
    float* g2c = (float*)alloc((size_t)Nnp[1] * 64 * 4);
    float* g1c = (float*)alloc((size_t)Nnp[0] * 64 * 4);
    float* sbuf = (float*)alloc((size_t)Nnp[0] * KB * 96 * 4);
    float* accb = (float*)alloc((size_t)Nnp[0] * 64 * 4);
    float* rootb = (float*)alloc((size_t)Nnp[0] * 64 * 4);
    unsigned* enc = (unsigned*)alloc((size_t)Nnp[0] * 64 * 4);
    float* logits = (float*)alloc((size_t)Nnp[0] * 4 * 4);
    // packed f16 weights
    half_t *cwsw[10], *crsw[10], *fc1sw, *fc2sw, *sksw[3];
    for (int i = 0; i < 10; ++i) {
        int kb = pad32(cinA[i]) / 32, nb = (coutA[i] + 15) / 16;
        cwsw[i] = (half_t*)alloc((size_t)K3 * kb * nb * 512 * 2);
        crsw[i] = (half_t*)alloc((size_t)kb * nb * 512 * 2);
    }
    fc1sw = (half_t*)alloc((size_t)2 * 4 * 512 * 2);
    fc2sw = (half_t*)alloc((size_t)2 * 1 * 512 * 2);
    for (int i = 0; i < 3; ++i) sksw[i] = (half_t*)alloc((size_t)2 * 1 * 512 * 2);

    // ---- pre-pack all B operands to swizzled f16 ----
    for (int i = 0; i < 10; ++i) {
        pack_b(stream, cw[i], cwsw[i], cinA[i], coutA[i], K3);
        pack_b(stream, cr[i], crsw[i], cinA[i], coutA[i], 1);
    }
    pack_b(stream, fc1_w, fc1sw, 64, 64, 1);
    pack_b(stream, fc2_w, fc2sw, 64, 4, 1);
    for (int i = 0; i < 3; ++i) pack_b(stream, sk_w[i], sksw[i], 64, 16, 1);

    // ---- basis + degree per level; pad x ----
    for (int l = 0; l < 4; ++l) {
        basis_kernel<<<ceil_div(En[l], 256), 256, 0, stream>>>(pseudo[l], basis[l], widx[l], En[l]);
        zero_f32<<<ceil_div(Nn[l], 256), 256, 0, stream>>>(deg[l], Nn[l]);
        deg_kernel<<<ceil_div(En[l], 256), 256, 0, stream>>>(ei[l] + En[l], deg[l], En[l]);
    }
    xpad_kernel<<<ceil_div(N1 * 32, 256), 256, 0, stream>>>(x, xpad, N1);

    // ---- encoder ----
    run_conv(stream, xpad, 32, N1, Nnp[0], 1, 32, 32, ei[0], ei[0]+E1, basis[0], widx[0], E1,
             deg[0], cwsw[0], crsw[0], cb[0], tmpA, 32, 0, sbuf, accb, rootb);
    run_conv(stream, tmpA, 32, N1, Nnp[0], 32, 32, 64, ei[0], ei[0]+E1, basis[0], widx[0], E1,
             deg[0], cwsw[1], crsw[1], cb[1], h1, 64, 0, sbuf, accb, rootb);
    run_pool(stream, h1, cl[0], N1, N2, 64, tmpA, enc);
    run_conv(stream, tmpA, 64, N2, Nnp[1], 64, 64, 64, ei[1], ei[1]+E2, basis[1], widx[1], E2,
             deg[1], cwsw[2], crsw[2], cb[2], tmpB, 64, 0, sbuf, accb, rootb);
    run_conv(stream, tmpB, 64, N2, Nnp[1], 64, 64, 64, ei[1], ei[1]+E2, basis[1], widx[1], E2,
             deg[1], cwsw[3], crsw[3], cb[3], h2, 64, 0, sbuf, accb, rootb);
    run_pool(stream, h2, cl[1], N2, N3, 64, tmpA, enc);
    run_conv(stream, tmpA, 64, N3, Nnp[2], 64, 64, 64, ei[2], ei[2]+E3, basis[2], widx[2], E3,
             deg[2], cwsw[4], crsw[4], cb[4], tmpB, 64, 0, sbuf, accb, rootb);
    run_conv(stream, tmpB, 64, N3, Nnp[2], 64, 64, 64, ei[2], ei[2]+E3, basis[2], widx[2], E3,
             deg[2], cwsw[5], crsw[5], cb[5], h3, 64, 0, sbuf, accb, rootb);
    run_pool(stream, h3, cl[2], N3, N4, 64, tmpA, enc);
    run_conv(stream, tmpA, 64, N4, Nnp[3], 64, 64, 64, ei[3], ei[3]+E4, basis[3], widx[3], E4,
             deg[3], cwsw[6], crsw[6], cb[6], tmpB, 64, 0, sbuf, accb, rootb);
    // fc1 + elu
    launch_gemm(stream, Nnp[3] / 16, tmpB, 64, fc1sw, fc1_b, h4, 64, 64, 4, 64, 0, 3);

    // ---- decoder with skip connections ----
    zero_f32<<<ceil_div(Nnp[2] * 96, 256), 256, 0, stream>>>(g3, (long long)Nnp[2] * 96);
    gather_rows<<<ceil_div(N3 * 64, 256), 256, 0, stream>>>(h4, 64, cl[2], g3, N3, 64, 96);
    launch_gemm(stream, Nnp[2] / 16, h3, 64, sksw[2], sk_b[2], g3, 64, 16, 1, 96, 64, 1);
    run_conv(stream, g3, 96, N3, Nnp[2], 80, 96, 64, ei[2], ei[2]+E3, basis[2], widx[2], E3,
             deg[2], cwsw[7], crsw[7], cb[7], g3c, 64, 0, sbuf, accb, rootb);

    zero_f32<<<ceil_div(Nnp[1] * 96, 256), 256, 0, stream>>>(g2, (long long)Nnp[1] * 96);
    gather_rows<<<ceil_div(N2 * 64, 256), 256, 0, stream>>>(g3c, 64, cl[1], g2, N2, 64, 96);
    launch_gemm(stream, Nnp[1] / 16, h2, 64, sksw[1], sk_b[1], g2, 64, 16, 1, 96, 64, 1);
    run_conv(stream, g2, 96, N2, Nnp[1], 80, 96, 64, ei[1], ei[1]+E2, basis[1], widx[1], E2,
             deg[1], cwsw[8], crsw[8], cb[8], g2c, 64, 0, sbuf, accb, rootb);

    zero_f32<<<ceil_div(Nnp[0] * 96, 256), 256, 0, stream>>>(g1, (long long)Nnp[0] * 96);
    gather_rows<<<ceil_div(N1 * 64, 256), 256, 0, stream>>>(g2c, 64, cl[0], g1, N1, 64, 96);
    launch_gemm(stream, Nnp[0] / 16, h1, 64, sksw[0], sk_b[0], g1, 64, 16, 1, 96, 64, 1);
    run_conv(stream, g1, 96, N1, Nnp[0], 80, 96, 64, ei[0], ei[0]+E1, basis[0], widx[0], E1,
             deg[0], cwsw[9], crsw[9], cb[9], g1c, 64, 0, sbuf, accb, rootb);

    // fc2 + elu, then log_softmax
    launch_gemm(stream, Nnp[0] / 16, g1c, 64, fc2sw, fc2_b, logits, 64, 4, 1, 4, 0, 3);
    log_softmax4<<<ceil_div(N1, 256), 256, 0, stream>>>(logits, (float*)d_out, N1);
}
